// interpolate_xy_33655363731579
// MI455X (gfx1250) — matrix operand order repaired and checked by reading, not compiled, not run on hardware
//
#include <hip/hip_runtime.h>
#include <hip/hip_bf16.h>

typedef _Float16 v16h __attribute__((ext_vector_type(16)));
typedef _Float16 v8h  __attribute__((ext_vector_type(8)));
typedef _Float16 h2   __attribute__((ext_vector_type(2)));
typedef float    v8f  __attribute__((ext_vector_type(8)));

#define NBATCH 2
#define QN     131072
#define HH     64
#define WW     2048
#define CC     64

// ---------------- weight transpose + f16 convert: dst[n*Kpad+k] = w[k*dout+n] ----------------
__global__ void wconv_kernel(const float* __restrict__ w, _Float16* __restrict__ dst,
                             int din, int dout, int Kpad, int Npad) {
    int idx = blockIdx.x * blockDim.x + threadIdx.x;
    int total = Kpad * Npad;
    if (idx >= total) return;
    int n = idx / Kpad;
    int k = idx - n * Kpad;
    float v = (k < din && n < dout) ? w[k * dout + n] : 0.0f;
    dst[(size_t)n * Kpad + k] = (_Float16)v;
}

// ---------------- WMMA fragment helpers (wave32, 16x16x32 f16) ----------------
// A (16x32 f16): lane r=l&15 is row; halves 0..15 = K[hi*16 .. hi*16+15]
__device__ __forceinline__ v16h load_a(const _Float16* buf, int stride, int kb, int lane) {
    int r = lane & 15, hi = lane >> 4;
    const _Float16* p = buf + r * stride + kb + hi * 16;
    union { v16h v; v8h h[2]; } u;
    u.h[0] = *(const v8h*)(p);
    u.h[1] = *(const v8h*)(p + 8);
    return u.v;
}
// B (32x16 f16): lane holds column n = nb+(l&15); 16 contiguous halves K = kb + hi*16 .. +15
// weights pre-transposed: wt[n*Kpad + k]
__device__ __forceinline__ v16h load_b(const _Float16* wt, int Kpad, int nb, int kb, int lane) {
    int n = nb + (lane & 15), hi = lane >> 4;
    return *(const v16h*)(wt + (size_t)n * Kpad + kb + hi * 16);
}
__device__ __forceinline__ v8f wmma_f16(v16h a, v16h b, v8f c) {
    return __builtin_amdgcn_wmma_f32_16x16x32_f16(false, a, false, b, (short)0, c, false, false);
}

// Generic dense layer: out[16][N] = act(in[16][K] * Wt + b).  A fragments hoisted to
// registers first so in==out aliasing (in-place layers) is safe (LDS in-order per wave).
template<int K, int N, bool RELU>
__device__ __forceinline__ void mlp_layer(const _Float16* in, int in_stride,
                                          _Float16* out, int out_stride,
                                          const _Float16* wt, const float* bias, int lane) {
    constexpr int KT = K / 32;
    v16h a[KT];
    #pragma unroll
    for (int k = 0; k < KT; ++k) a[k] = load_a(in, in_stride, k * 32, lane);
    int nl = lane & 15, hi = lane >> 4;
    #pragma unroll
    for (int nb = 0; nb < N; nb += 16) {
        v8f c = {0.f, 0.f, 0.f, 0.f, 0.f, 0.f, 0.f, 0.f};
        #pragma unroll
        for (int k = 0; k < KT; ++k)
            c = wmma_f16(a[k], load_b(wt, K, nb, k * 32, lane), c);
        float bv = bias[nb + nl];
        #pragma unroll
        for (int i = 0; i < 8; ++i) {
            float v = c[i] + bv;
            if (RELU) v = fmaxf(v, 0.f);
            out[(hi * 8 + i) * out_stride + nb + nl] = (_Float16)v;
        }
    }
}

// ---------------- fused gather + PE-MLP + encoder-MLP + softmax kernel ----------------
__global__ void __launch_bounds__(128)
fused_kernel(const float* __restrict__ depth, const float* __restrict__ feat,
             const float* __restrict__ coord,
             const float* __restrict__ pe_b0, const float* __restrict__ pe_b1,
             const float* __restrict__ pe_b2,
             const float* __restrict__ e_b0, const float* __restrict__ e_b1,
             const float* __restrict__ e_b2, const float* __restrict__ e_b3,
             const _Float16* __restrict__ wpe0, const _Float16* __restrict__ wpe1,
             const _Float16* __restrict__ wpe2,
             const _Float16* __restrict__ we0, const _Float16* __restrict__ we1,
             const _Float16* __restrict__ we2, const _Float16* __restrict__ we3,
             float* __restrict__ out)
{
    __shared__ _Float16 s_qf[4][16 * 256];   // per-wave qf / enc activations
    __shared__ _Float16 s_tmp[4][16 * 128];  // per-wave PE/enc intermediates (+logit scratch)
    __shared__ float    s_rc[4][16][8];
    __shared__ float    s_p [4][16][4];

    const int wv   = threadIdx.x >> 5;
    const int lane = threadIdx.x & 31;
    const int rl   = lane & 15;
    const int hi   = lane >> 4;
    const int rowBase = blockIdx.x * 64 + wv * 16;
    const int gr = rowBase + rl;              // global row (b*Q + qi)
    const int b  = gr >> 17;                  // Q = 131072 = 2^17

    _Float16* qf = s_qf[wv];
    _Float16* tb = s_tmp[wv];

    const float cy = coord[2 * gr + 0];
    const float cx = coord[2 * gr + 1];

    const float offy[4] = {-1.f, -1.f, 1.f, 1.f};
    const float offx[4] = {-1.f,  1.f, -1.f, 1.f};

    // ---- gather stage: each lane handles 2 of this row's 4 corners ----
    #pragma unroll
    for (int j = 0; j < 2; ++j) {
        int k = hi * 2 + j;
        float sy = offy[k] * (1.f / HH) + 1e-6f;
        float sx = offx[k] * (1.f / WW) + 1e-6f;
        float cyy = fminf(fmaxf(cy + sy, -1.f + 1e-6f), 1.f - 1e-6f);
        float cxx = fminf(fmaxf(cx + sx, -1.f + 1e-6f), 1.f - 1e-6f);
        int iy = (int)rintf(((cyy + 1.f) * HH - 1.f) * 0.5f);  // RNE == jnp.round
        iy = min(max(iy, 0), HH - 1);
        int ix = (int)rintf(((cxx + 1.f) * WW - 1.f) * 0.5f);
        ix = min(max(ix, 0), WW - 1);

        const size_t S = (size_t)HH * WW;
        const float* fb = feat + (size_t)b * CC * S + (size_t)iy * WW + ix;
        _Float16* dst = qf + rl * 256 + k * 64;
        for (int c = 0; c < CC; c += 2) {
            float f0 = fb[(size_t)c * S];
            float f1 = fb[(size_t)(c + 1) * S];
            h2 hv = { (_Float16)f0, (_Float16)f1 };
            *(h2*)(dst + c) = hv;
        }
        float d1 = depth[(((size_t)b * 3 + 1) * HH + iy) * WW + ix];
        float d2 = depth[(((size_t)b * 3 + 2) * HH + iy) * WW + ix];
        s_p[wv][rl][k] = sqrtf(d1 * d1 + d2 * d2);
        float qcy = -1.f + (2.f * iy + 1.f) / HH;
        float qcx = -1.f + (2.f * ix + 1.f) / WW;
        s_rc[wv][rl][2 * k + 0] = (cy - qcy) * HH;
        s_rc[wv][rl][2 * k + 1] = (cx - qcx) * WW;
    }
    asm volatile("s_wait_dscnt 0x0" ::: "memory");

    // ---- PE L0: rc (K=8, zero-padded to 32) -> 64, relu ----
    {
        union { v16h v; _Float16 e[16]; } a;
        #pragma unroll
        for (int i = 0; i < 16; ++i) a.e[i] = (_Float16)0.f;
        if (hi == 0) {
            #pragma unroll
            for (int j = 0; j < 8; ++j) a.e[j] = (_Float16)s_rc[wv][rl][j];
        }
        #pragma unroll
        for (int nb = 0; nb < 64; nb += 16) {
            v8f c = {0.f, 0.f, 0.f, 0.f, 0.f, 0.f, 0.f, 0.f};
            c = wmma_f16(a.v, load_b(wpe0, 32, nb, 0, lane), c);
            float bv = pe_b0[nb + rl];
            #pragma unroll
            for (int i = 0; i < 8; ++i)
                tb[(hi * 8 + i) * 128 + nb + rl] = (_Float16)fmaxf(c[i] + bv, 0.f);
        }
    }
    // ---- PE L1: 64 -> 128, relu (in-place in tb; A hoisted) ----
    mlp_layer<64, 128, true>(tb, 128, tb, 128, wpe1, pe_b1, lane);
    // ---- PE L2: 128 -> 256, add to qf, no relu ----
    {
        v16h a[4];
        #pragma unroll
        for (int k = 0; k < 4; ++k) a[k] = load_a(tb, 128, k * 32, lane);
        #pragma unroll
        for (int nb = 0; nb < 256; nb += 16) {
            v8f c = {0.f, 0.f, 0.f, 0.f, 0.f, 0.f, 0.f, 0.f};
            #pragma unroll
            for (int k = 0; k < 4; ++k)
                c = wmma_f16(a[k], load_b(wpe2, 128, nb, k * 32, lane), c);
            float bv = pe_b2[nb + rl];
            #pragma unroll
            for (int i = 0; i < 8; ++i) {
                int idx = (hi * 8 + i) * 256 + nb + rl;
                qf[idx] = (_Float16)(c[i] + bv + (float)qf[idx]);
            }
        }
    }
    // ---- encoder ----
    mlp_layer<256, 256, true>(qf, 256, qf, 256, we0, e_b0, lane);  // in-place, A hoisted
    mlp_layer<256, 128, true>(qf, 256, tb, 128, we1, e_b1, lane);
    mlp_layer<128,  64, true>(tb, 128, qf, 256, we2, e_b2, lane);
    // ---- enc L3: 64 -> 4 (N padded to 16); f32 logits to scratch ----
    float* logits = (float*)tb;  // reused as [16][4] f32
    {
        v16h a0 = load_a(qf, 256, 0, lane);
        v16h a1 = load_a(qf, 256, 32, lane);
        v8f c = {0.f, 0.f, 0.f, 0.f, 0.f, 0.f, 0.f, 0.f};
        c = wmma_f16(a0, load_b(we3, 64, 0, 0, lane), c);
        c = wmma_f16(a1, load_b(we3, 64, 0, 32, lane), c);
        if (rl < 4) {
            float bv = e_b3[rl];
            #pragma unroll
            for (int i = 0; i < 8; ++i)
                logits[(hi * 8 + i) * 4 + rl] = c[i] + bv;
        }
    }
    asm volatile("s_wait_dscnt 0x0" ::: "memory");

    // ---- softmax over 4 + weighted pred_xy; one lane per row ----
    if (lane < 16) {
        float l0 = logits[lane * 4 + 0], l1 = logits[lane * 4 + 1];
        float l2 = logits[lane * 4 + 2], l3 = logits[lane * 4 + 3];
        float m = fmaxf(fmaxf(l0, l1), fmaxf(l2, l3));
        float e0 = expf(l0 - m), e1 = expf(l1 - m), e2 = expf(l2 - m), e3 = expf(l3 - m);
        float s = e0 + e1 + e2 + e3;
        float r = (s_p[wv][lane][0] * e0 + s_p[wv][lane][1] * e1 +
                   s_p[wv][lane][2] * e2 + s_p[wv][lane][3] * e3) / s;
        out[rowBase + lane] = r;
    }
}

// ---------------- launch ----------------
extern "C" void kernel_launch(void* const* d_in, const int* in_sizes, int n_in,
                              void* d_out, int out_size, void* d_ws, size_t ws_size,
                              hipStream_t stream) {
    const float* depth  = (const float*)d_in[0];
    const float* feat   = (const float*)d_in[1];
    const float* coord  = (const float*)d_in[2];
    const float* pe_w0  = (const float*)d_in[3];
    const float* pe_b0  = (const float*)d_in[4];
    const float* pe_w1  = (const float*)d_in[5];
    const float* pe_b1  = (const float*)d_in[6];
    const float* pe_w2  = (const float*)d_in[7];
    const float* pe_b2  = (const float*)d_in[8];
    const float* enc_w0 = (const float*)d_in[9];
    const float* enc_b0 = (const float*)d_in[10];
    const float* enc_w1 = (const float*)d_in[11];
    const float* enc_b1 = (const float*)d_in[12];
    const float* enc_w2 = (const float*)d_in[13];
    const float* enc_b2 = (const float*)d_in[14];
    const float* enc_w3 = (const float*)d_in[15];
    const float* enc_b3 = (const float*)d_in[16];

    _Float16* ws = (_Float16*)d_ws;
    // half offsets (all 32B-aligned)
    const size_t OFF_PE0 = 0;                        // 64  x 32  = 2048
    const size_t OFF_PE1 = OFF_PE0 + 64 * 32;        // 128 x 64  = 8192
    const size_t OFF_PE2 = OFF_PE1 + 128 * 64;       // 256 x 128 = 32768
    const size_t OFF_E0  = OFF_PE2 + 256 * 128;      // 256 x 256 = 65536
    const size_t OFF_E1  = OFF_E0  + 256 * 256;      // 128 x 256 = 32768
    const size_t OFF_E2  = OFF_E1  + 128 * 256;      // 64  x 128 = 8192
    const size_t OFF_E3  = OFF_E2  + 64 * 128;       // 16  x 64  = 1024

    auto conv = [&](const float* w, size_t off, int din, int dout, int Kp, int Np) {
        int total = Kp * Np;
        wconv_kernel<<<(total + 255) / 256, 256, 0, stream>>>(w, ws + off, din, dout, Kp, Np);
    };
    conv(pe_w0,  OFF_PE0,   8,  64,  32,  64);
    conv(pe_w1,  OFF_PE1,  64, 128,  64, 128);
    conv(pe_w2,  OFF_PE2, 128, 256, 128, 256);
    conv(enc_w0, OFF_E0,  256, 256, 256, 256);
    conv(enc_w1, OFF_E1,  256, 128, 256, 128);
    conv(enc_w2, OFF_E2,  128,  64, 128,  64);
    conv(enc_w3, OFF_E3,   64,   4,  64,  16);

    // 262144 rows / 64 rows-per-block
    fused_kernel<<<4096, 128, 0, stream>>>(
        depth, feat, coord,
        pe_b0, pe_b1, pe_b2, enc_b0, enc_b1, enc_b2, enc_b3,
        ws + OFF_PE0, ws + OFF_PE1, ws + OFF_PE2,
        ws + OFF_E0, ws + OFF_E1, ws + OFF_E2, ws + OFF_E3,
        (float*)d_out);
}
